// wide_basic_10522669875687
// MI455X (gfx1250) — compile-verified, matrix operand order checked by reading, not run on hardware
//
#include <hip/hip_runtime.h>

// ---- CDNA5 vector types -----------------------------------------------------
typedef __attribute__((ext_vector_type(16))) __bf16 v16bf;
typedef __attribute__((ext_vector_type(8)))  __bf16 v8bf;
typedef __attribute__((ext_vector_type(8)))  float  v8f;
typedef __attribute__((ext_vector_type(4)))  unsigned int u32x4;
typedef __attribute__((ext_vector_type(8)))  int          i32x8;
typedef __attribute__((ext_vector_type(4)))  int          i32x4;

union Frag {
    v16bf v;
    v8bf  h[2];
};

#define EPS 1e-5f
// Problem constants: x = (64, 256, 32, 32)
#define NB   64
#define CCH  256
#define HW   1024        // 32*32
#define NPOS (NB * HW)   // 65536 positions

#if defined(__HIP_DEVICE_COMPILE__) && __has_builtin(__builtin_amdgcn_tensor_load_to_lds)
#define USE_TDM 1
#else
#define USE_TDM 0
#endif

// ---------------------------------------------------------------------------
// Kernel 1: fused BN1 + ReLU + NCHW -> (N*H*W, C) transpose, f32 -> bf16
// ---------------------------------------------------------------------------
__global__ void prep_act_kernel(const float* __restrict__ x,
                                const float* __restrict__ g,
                                const float* __restrict__ b,
                                const float* __restrict__ mn,
                                const float* __restrict__ vr,
                                __bf16* __restrict__ a1) {
    int tid = blockIdx.x * 256 + threadIdx.x;     // one per element
    int c   = tid & 255;
    int pos = tid >> 8;
    int n   = pos >> 10;
    int hw  = pos & 1023;
    float inv  = g[c] * rsqrtf(vr[c] + EPS);
    float bias = b[c] - mn[c] * inv;
    float val  = x[((n << 8) + c) * HW + hw] * inv + bias;
    a1[tid] = (__bf16)fmaxf(val, 0.0f);
}

// ---------------------------------------------------------------------------
// Kernel 2: fold mask into weights, OIHW f32 -> [tap][Cout][Cin] bf16
// ---------------------------------------------------------------------------
__global__ void prep_w_kernel(const float* __restrict__ w,
                              const float* __restrict__ mask,
                              __bf16* __restrict__ wp) {
    int tid = blockIdx.x * 256 + threadIdx.x;     // t*65536 + o*256 + i
    int i   = tid & 255;
    int o   = (tid >> 8) & 255;
    int t   = tid >> 16;                          // 0..8 = kh*3+kw
    int src = (o * 256 + i) * 9 + t;
    wp[tid] = (__bf16)(w[src] * mask[src]);
}

// ---------------------------------------------------------------------------
// Implicit-GEMM 3x3 conv, pad=1, stride=1, via V_WMMA_F32_16X16X32_BF16.
// M = positions (N*H*W), N = out channels, K = 9 taps x 256 in-channels.
// Block = 256 threads = 8 wave32s; block tile 128(M) x 128(N);
// each wave owns a 64x32 subtile = 4x2 WMMA accumulators.
// B (weight) tile is DMA'd into LDS by the Tensor Data Mover (TENSORcnt).
// MODE 1: epilogue = BN2+ReLU -> bf16 activations
// MODE 2: epilogue = + residual x -> f32 NCHW output
// ---------------------------------------------------------------------------
template <int MODE>
__global__ __launch_bounds__(256)
void conv3x3_wmma_kernel(const __bf16* __restrict__ act,   // (NPOS, 256) bf16
                         const __bf16* __restrict__ wp,    // (9, 256, 256) bf16
                         const float* __restrict__ g2,
                         const float* __restrict__ b2,
                         const float* __restrict__ m2,
                         const float* __restrict__ v2,
                         const float* __restrict__ xres,   // NCHW f32 residual
                         __bf16* __restrict__ a2out,       // MODE 1 output
                         float* __restrict__ fout) {       // MODE 2 output
    __shared__ __bf16 As[128 * 32];   // 8 KB  (M-tile x k-chunk)
    __shared__ __bf16 Bs[128 * 32];   // 8 KB  (N-tile x k-chunk), TDM dest

    const int tid   = threadIdx.x;
    const int Mbase = blockIdx.x * 128;
    const int Nbase = blockIdx.y * 128;

    const int lane = tid & 31;
    const int wave = tid >> 5;
    const int wm   = wave & 1;    // 2 M-groups of 64 rows
    const int wn   = wave >> 1;   // 4 N-groups of 32 cols
    const int l15  = lane & 15;
    const int lhi  = lane >> 4;   // 0 or 1
    const int khalf = lhi << 3;   // K lane offset per ISA A/B bf16 layout

    // Cooperative A-tile coordinates: 128 rows x 2 half-chunks of 16 elems.
    const int ldr = tid >> 1;            // 0..127
    const int ldk = (tid & 1) << 4;      // 0 or 16

    v8f acc[4][2] = {};

    for (int tap = 0; tap < 9; ++tap) {
        const int dh = tap / 3 - 1;
        const int dw = tap % 3 - 1;

        // Shifted-source validity for this thread's A-tile row.
        const int pos = Mbase + ldr;
        const int hw  = pos & 1023;
        const int h   = hw >> 5;
        const int w   = hw & 31;
        const int h2  = h + dh;
        const int w2  = w + dw;
        const bool ok = ((unsigned)h2 < 32u) && ((unsigned)w2 < 32u);
        const int spos = pos + dh * 32 + dw;

        for (int kb = 0; kb < 8; ++kb) {
            const int kbase = kb << 5;   // 0,32,...,224
            __syncthreads();             // previous compute done, LDS free

            // ---- stage A tile (zero-padded at image borders) ----
            Frag fa = {};
            if (ok) {
                const __bf16* srcA = act + spos * 256 + kbase + ldk;
                fa.v = *(const v16bf*)srcA;
                if (kb < 7) __builtin_prefetch(srcA + 32, 0, 1);  // global_prefetch_b8
            }
            *(v8bf*)&As[ldr * 32 + ldk]     = fa.h[0];
            *(v8bf*)&As[ldr * 32 + ldk + 8] = fa.h[1];

            // ---- stage B tile via Tensor Data Mover ----
#if USE_TDM
            if (tid < 32) {   // one wave issues one TDM op (EXEC ignored)
                const __bf16* tileB = wp + ((tap << 8) + Nbase) * 256 + kbase;
                unsigned long long ga = (unsigned long long)(const void*)tileB;
                unsigned lds_off = (unsigned)(unsigned long long)(const void*)&Bs[0];

                // D# group 0: count=1 | lds_addr | global_addr | type=2
                u32x4 g0;
                g0[0] = 1u;
                g0[1] = lds_off;
                g0[2] = (unsigned)(ga & 0xffffffffull);
                g0[3] = (unsigned)((ga >> 32) & 0x01ffffffull) | (2u << 30);

                // D# group 1: data_size=2B; tensor_dim0=256, tensor_dim1=4096;
                //             tile = 32(k) x 128(cout); dim0_stride=256.
                i32x8 g1;
                g1[0] = 0x00010000;            // wg_mask=0, data_size=1 (2B)
                g1[1] = (int)(256u << 16);     // tensor_dim0[15:0] in [31:16]
                g1[2] = (int)(4096u << 16);    // dim0 hi=0, tensor_dim1 lo=4096
                g1[3] = (int)(32u  << 16);     // dim1 hi=0, tile_dim0=32
                g1[4] = 128;                   // tile_dim1=128, tile_dim2=0
                g1[5] = 256;                   // tensor_dim0_stride lo
                g1[6] = 0;                     // stride0 hi, stride1 lo
                g1[7] = 0;                     // stride1 hi
                i32x4 z4 = {};
#if __clang_major__ >= 23
                i32x8 z8 = {};
                __builtin_amdgcn_tensor_load_to_lds(g0, g1, z4, z4, z8, 0);
#else
                __builtin_amdgcn_tensor_load_to_lds(g0, g1, z4, z4, 0);
#endif
                __builtin_amdgcn_s_wait_tensorcnt(0);
            }
#else
            {   // fallback: cooperative vector-load staging of B
                Frag fb;
                const __bf16* srcB =
                    wp + ((tap << 8) + Nbase + ldr) * 256 + kbase + ldk;
                fb.v = *(const v16bf*)srcB;
                *(v8bf*)&Bs[ldr * 32 + ldk]     = fb.h[0];
                *(v8bf*)&Bs[ldr * 32 + ldk + 8] = fb.h[1];
            }
#endif
            __syncthreads();   // publishes A (ds) and B (TDM) tiles

            // ---- fragments per ISA 7.12.2 (16-bit A 16x32 wave32 layout) ----
            v16bf afr[4];
            v16bf bfr[2];
#pragma unroll
            for (int ms = 0; ms < 4; ++ms) {
                const int row = (wm << 6) + (ms << 4) + l15;
                Frag f;
                f.h[0] = *(const v8bf*)&As[row * 32 + khalf];        // K khalf..+7
                f.h[1] = *(const v8bf*)&As[row * 32 + khalf + 16];   // K khalf+16..+23
                afr[ms] = f.v;
            }
#pragma unroll
            for (int ns = 0; ns < 2; ++ns) {
                const int col = (wn << 5) + (ns << 4) + l15;
                Frag f;
                f.h[0] = *(const v8bf*)&Bs[col * 32 + khalf];
                f.h[1] = *(const v8bf*)&Bs[col * 32 + khalf + 16];
                bfr[ns] = f.v;
            }

            // ---- 8 WMMAs: 4x2 subtiles of 16x16, K=32 ----
#pragma unroll
            for (int ms = 0; ms < 4; ++ms)
#pragma unroll
                for (int ns = 0; ns < 2; ++ns)
                    acc[ms][ns] = __builtin_amdgcn_wmma_f32_16x16x32_bf16(
                        false, afr[ms], false, bfr[ns],
                        (short)0, acc[ms][ns], false, false);
        }
    }

    // ---- epilogue: C/D layout (lane -> col l15, VGPR r -> row r + 8*lhi) ----
#pragma unroll
    for (int ms = 0; ms < 4; ++ms) {
#pragma unroll
        for (int ns = 0; ns < 2; ++ns) {
            const int col = Nbase + (wn << 5) + (ns << 4) + l15;
            float inv = 0.0f, bias = 0.0f;
            if (MODE == 1) {
                inv  = g2[col] * rsqrtf(v2[col] + EPS);
                bias = b2[col] - m2[col] * inv;
            }
#pragma unroll
            for (int r = 0; r < 8; ++r) {
                const int row = Mbase + (wm << 6) + (ms << 4) + r + (lhi << 3);
                float val = acc[ms][ns][r];
                if (MODE == 1) {
                    val = fmaxf(val * inv + bias, 0.0f);
                    a2out[row * 256 + col] = (__bf16)val;
                } else {
                    const int n   = row >> 10;
                    const int hw2 = row & 1023;
                    const int idx = ((n << 8) + col) * HW + hw2;
                    fout[idx] = val + xres[idx];
                }
            }
        }
    }
}

// ---------------------------------------------------------------------------
// Host launcher
// ---------------------------------------------------------------------------
extern "C" void kernel_launch(void* const* d_in, const int* in_sizes, int n_in,
                              void* d_out, int out_size, void* d_ws, size_t ws_size,
                              hipStream_t stream) {
    (void)in_sizes; (void)n_in; (void)out_size; (void)ws_size;

    const float* x    = (const float*)d_in[0];
    const float* g1   = (const float*)d_in[1];
    const float* b1   = (const float*)d_in[2];
    const float* m1   = (const float*)d_in[3];
    const float* v1   = (const float*)d_in[4];
    const float* w1   = (const float*)d_in[5];
    const float* msk1 = (const float*)d_in[6];
    const float* g2   = (const float*)d_in[7];
    const float* b2   = (const float*)d_in[8];
    const float* m2   = (const float*)d_in[9];
    const float* v2   = (const float*)d_in[10];
    const float* w2   = (const float*)d_in[11];
    const float* msk2 = (const float*)d_in[12];
    float* out = (float*)d_out;

    // Workspace layout (bytes):
    //   a1  : NPOS*256 bf16 = 33 554 432
    //   a2  : NPOS*256 bf16 = 33 554 432
    //   wp1 : 9*256*256 bf16 =  1 179 648
    //   wp2 : 9*256*256 bf16 =  1 179 648
    char* ws = (char*)d_ws;
    __bf16* a1  = (__bf16*)(ws);
    __bf16* a2  = (__bf16*)(ws + 33554432u);
    __bf16* wp1 = (__bf16*)(ws + 67108864u);
    __bf16* wp2 = (__bf16*)(ws + 68288512u);

    // 1) BN1 + ReLU + transpose -> bf16 activations
    prep_act_kernel<<<NPOS * CCH / 256, 256, 0, stream>>>(x, g1, b1, m1, v1, a1);
    // 2) masked weights -> [tap][Cout][Cin] bf16
    prep_w_kernel<<<9 * CCH * CCH / 256, 256, 0, stream>>>(w1, msk1, wp1);
    prep_w_kernel<<<9 * CCH * CCH / 256, 256, 0, stream>>>(w2, msk2, wp2);

    // 3) conv1 + fused BN2+ReLU -> bf16 a2 ; 4) conv2 + residual -> f32 out
    dim3 grid(NPOS / 128, CCH / 128);   // 512 x 2
    conv3x3_wmma_kernel<1><<<grid, 256, 0, stream>>>(
        a1, wp1, g2, b2, m2, v2, nullptr, a2, nullptr);
    conv3x3_wmma_kernel<2><<<grid, 256, 0, stream>>>(
        a2, wp2, nullptr, nullptr, nullptr, nullptr, x, nullptr, out);
}
